// Decoder_55087250538592
// MI455X (gfx1250) — matrix operand order repaired and checked by reading, not compiled, not run on hardware
//
#include <hip/hip_runtime.h>
#include <math.h>

// ---------------------------------------------------------------------------
// CDNA5 (gfx1250) fused decoder, single-pass flash-style attention.
// One workgroup (256 thr = 8 waves) per batch row; wave w == head w.
// K/V projections run as interleaved v_wmma_f32_16x16x32_bf16 chains.
// ---------------------------------------------------------------------------

typedef __bf16 bf16_t;
typedef __attribute__((ext_vector_type(16))) bf16_t v16bf;
typedef __attribute__((ext_vector_type(8)))  bf16_t v8bf;
typedef __attribute__((ext_vector_type(8)))  float  v8f;

#define E_DIM 128
#define T_DIM 512
#define H_DIM 8
#define NTILE 32   // T / 16

__device__ __forceinline__ unsigned short f2bf(float f) {
    union { float f; unsigned u; } x; x.f = f;
    unsigned r = x.u + 0x7fffu + ((x.u >> 16) & 1u);   // round-to-nearest-even
    return (unsigned short)(r >> 16);
}

// LayerNorm of a 128-vector in LDS -> dst (LDS). red = 2-float scratch.
__device__ __forceinline__ void ln_vec(const float* src, const float* g, const float* bb,
                                       float* dst, float* red, int tid) {
    if (tid < 32) {
        float s = 0.f, s2 = 0.f;
#pragma unroll
        for (int k = 0; k < 4; ++k) { float v = src[tid + 32 * k]; s += v; s2 += v * v; }
#pragma unroll
        for (int m = 1; m <= 16; m <<= 1) { s += __shfl_xor(s, m, 32); s2 += __shfl_xor(s2, m, 32); }
        if (tid == 0) {
            float mean = s * (1.f / 128.f);
            float var  = s2 * (1.f / 128.f) - mean * mean;
            red[0] = mean; red[1] = rsqrtf(var + 1e-5f);
        }
    }
    __syncthreads();
    if (tid < 128) dst[tid] = (src[tid] - red[0]) * red[1] * g[tid] + bb[tid];
}

// One LN-statistics pass over a 16x128 enc tile, emitting BOTH ln1- and
// ln2-scaled bf16 tiles (kx -> sXa, vx -> sXb). gamma/beta pre-staged in regs.
__device__ __forceinline__ void ln_tile2(const float* __restrict__ encb, int t0,
                                         const float* g1r, const float* b1r,
                                         const float* g2r, const float* b2r,
                                         unsigned short* sXa, unsigned short* sXb,
                                         float* red, float* mv, int tid) {
    const int row = tid >> 4;
    const int c0  = (tid & 15) * 8;
    const float4* p = (const float4*)(encb + (size_t)(t0 + row) * E_DIM + c0);
    float4 u0 = p[0], u1 = p[1];
    float x[8] = { u0.x, u0.y, u0.z, u0.w, u1.x, u1.y, u1.z, u1.w };
    float s = 0.f, s2 = 0.f;
#pragma unroll
    for (int j = 0; j < 8; ++j) { s += x[j]; s2 += x[j] * x[j]; }
    red[tid] = s; red[256 + tid] = s2;
    __syncthreads();
    if ((tid & 15) == 0) {
        float ts = 0.f, ts2 = 0.f;
#pragma unroll
        for (int k = 0; k < 16; ++k) { ts += red[row * 16 + k]; ts2 += red[256 + row * 16 + k]; }
        float mean = ts * (1.f / 128.f);
        float var  = ts2 * (1.f / 128.f) - mean * mean;
        mv[row] = mean; mv[16 + row] = rsqrtf(var + 1e-5f);
    }
    __syncthreads();
    const float mean = mv[row], rs = mv[16 + row];
#pragma unroll
    for (int j = 0; j < 8; ++j) {
        float v = (x[j] - mean) * rs;
        sXa[row * E_DIM + c0 + j] = f2bf(v * g1r[j] + b1r[j]);
        sXb[row * E_DIM + c0 + j] = f2bf(v * g2r[j] + b2r[j]);
    }
}

__global__ __launch_bounds__(256) void decoder_kernel(
    const float* __restrict__ enc, const float* __restrict__ q0,
    const float* __restrict__ wk, const float* __restrict__ wq, const float* __restrict__ wv,
    const float* __restrict__ projw, const float* __restrict__ projb,
    const float* __restrict__ ln1g, const float* __restrict__ ln1b,
    const float* __restrict__ ln2g, const float* __restrict__ ln2b,
    const float* __restrict__ ln3g, const float* __restrict__ ln3b,
    const float* __restrict__ ln4g, const float* __restrict__ ln4b,
    const float* __restrict__ ffw1, const float* __restrict__ ffb1,
    const float* __restrict__ ffw2, const float* __restrict__ ffb2,
    const float* __restrict__ lnfg, const float* __restrict__ lnfb,
    const float* __restrict__ headw, const float* __restrict__ headb,
    float* __restrict__ out) {
    extern __shared__ __align__(32) char dynLds[];
    unsigned short* sWTv = (unsigned short*)dynLds;                            // wv^T bf16 (32KB dyn)
    __shared__ __attribute__((aligned(32))) unsigned short sWTk[E_DIM * E_DIM]; // wk^T bf16 (32KB)
    __shared__ __attribute__((aligned(32))) unsigned short sXa[16 * E_DIM];     // ln1 tile bf16
    __shared__ __attribute__((aligned(32))) unsigned short sXb[16 * E_DIM];     // ln2 tile bf16
    __shared__ __attribute__((aligned(16))) float sWei[H_DIM * 16];             // per-tile softmax wts
    __shared__ float sFfn[640];     // hx[128] + h512[512]
    __shared__ float sQ[E_DIM];
    __shared__ float sq[E_DIM];
    __shared__ float sAtt[E_DIM];
    __shared__ float sTmp[512];
    __shared__ float sMV[32];
    __shared__ float sRed[2];

    const int tid  = threadIdx.x;
    const int lane = tid & 31;
    const int w    = tid >> 5;                 // wave id == head id
    const int b    = blockIdx.x;
    const float* encb = enc + (size_t)b * T_DIM * E_DIM;
    const int rowp = tid >> 4;
    const int c0p  = (tid & 15) * 8;

    if (tid < E_DIM) sq[tid] = q0[(size_t)b * E_DIM + tid];
    __syncthreads();

    for (int l = 0; l < 3; ++l) {
        const float* WK  = wk    + (size_t)l * E_DIM * E_DIM;
        const float* WQ  = wq    + (size_t)l * E_DIM * E_DIM;
        const float* WV  = wv    + (size_t)l * E_DIM * E_DIM;
        const float* PW  = projw + (size_t)l * E_DIM * E_DIM;
        const float* PB  = projb + (size_t)l * E_DIM;
        const float* F1W = ffw1  + (size_t)l * E_DIM * 512;
        const float* F1B = ffb1  + (size_t)l * 512;
        const float* F2W = ffw2  + (size_t)l * 512 * E_DIM;
        const float* F2B = ffb2  + (size_t)l * E_DIM;

        // ---- Q = ln3(q) @ wq ----
        ln_vec(sq, ln3g + l * E_DIM, ln3b + l * E_DIM, sTmp, sRed, tid);
        __syncthreads();
        for (int j = tid; j < E_DIM; j += 256) {
            float acc = 0.f;
            for (int i = 0; i < E_DIM; ++i) acc += sTmp[i] * WQ[i * E_DIM + j];
            sQ[j] = acc;
        }
        __syncthreads();

        // ---- stage wk^T and wv^T into LDS as bf16 (float4 loads) ----
        for (int idx = tid; idx < (E_DIM * E_DIM) / 4; idx += 256) {
            const int k  = idx >> 5;
            const int n4 = (idx & 31) * 4;
            float4 a = ((const float4*)WK)[idx];
            sWTk[(n4 + 0) * E_DIM + k] = f2bf(a.x);
            sWTk[(n4 + 1) * E_DIM + k] = f2bf(a.y);
            sWTk[(n4 + 2) * E_DIM + k] = f2bf(a.z);
            sWTk[(n4 + 3) * E_DIM + k] = f2bf(a.w);
            float4 vv = ((const float4*)WV)[idx];
            sWTv[(n4 + 0) * E_DIM + k] = f2bf(vv.x);
            sWTv[(n4 + 1) * E_DIM + k] = f2bf(vv.y);
            sWTv[(n4 + 2) * E_DIM + k] = f2bf(vv.z);
            sWTv[(n4 + 3) * E_DIM + k] = f2bf(vv.w);
        }
        __syncthreads();

        // ---- per-thread LN params in registers ----
        float g1r[8], b1r[8], g2r[8], b2r[8];
#pragma unroll
        for (int j = 0; j < 8; ++j) {
            g1r[j] = ln1g[l * E_DIM + c0p + j]; b1r[j] = ln1b[l * E_DIM + c0p + j];
            g2r[j] = ln2g[l * E_DIM + c0p + j]; b2r[j] = ln2b[l * E_DIM + c0p + j];
        }
        // reference MULTIPLIES logits by sqrt(D)=4 -- fold into Q
        const float qd4 = sQ[w * 16 + (lane & 15)] * 4.0f;

        // ---- single pass over T: K-proj -> scores -> online softmax -> V acc ----
        float mrun = -1e30f, srun = 0.f, acc = 0.f;
        for (int tt = 0; tt < NTILE; ++tt) {
            const int t0 = tt * 16;
            if (tt + 1 < NTILE)   // software prefetch of next enc tile
                __builtin_prefetch(encb + (size_t)(t0 + 16 + rowp) * E_DIM + c0p, 0, 3);
            ln_tile2(encb, t0, g1r, b1r, g2r, b2r, sXa, sXb, sTmp, sMV, tid);
            __syncthreads();

            // interleaved K and V WMMA chains (independent accumulators)
            v8f ck = {}, cv = {};
            {
                const int m     = lane & 15;
                const int apart = (lane >> 4) << 3;
                const int npart = (lane >> 4) << 4;
                const int n     = w * 16 + (lane & 15);
#pragma unroll
                for (int s = 0; s < 4; ++s) {
                    union { v16bf v; v8bf h[2]; } ak, av;
                    ak.h[0] = *(const v8bf*)(sXa + m * E_DIM + 32 * s + apart);
                    ak.h[1] = *(const v8bf*)(sXa + m * E_DIM + 32 * s + 16 + apart);
                    av.h[0] = *(const v8bf*)(sXb + m * E_DIM + 32 * s + apart);
                    av.h[1] = *(const v8bf*)(sXb + m * E_DIM + 32 * s + 16 + apart);
                    union { v16bf v; v8bf h[2]; } bk, bv;
                    bk.h[0] = *(const v8bf*)(sWTk + n * E_DIM + 32 * s + apart);
                    bk.h[1] = *(const v8bf*)(sWTk + n * E_DIM + 32 * s + 16 + apart);
                    bv.h[0] = *(const v8bf*)(sWTv + n * E_DIM + 32 * s + apart);
                    bv.h[1] = *(const v8bf*)(sWTv + n * E_DIM + 32 * s + 16 + apart);
                    ck = __builtin_amdgcn_wmma_f32_16x16x32_bf16(false, ak.v, false, bk.v, (short)0, ck, false, false);
                    cv = __builtin_amdgcn_wmma_f32_16x16x32_bf16(false, av.v, false, bv.v, (short)0, cv, false, false);
                }
            }

            // compacting tree reduce: 8 row-partials x 16 lanes -> 1 row total/lane
            float p0[8];
#pragma unroll
            for (int r = 0; r < 8; ++r) p0[r] = ck[r] * qd4;
            float q4[4];
#pragma unroll
            for (int r = 0; r < 4; ++r) {
                float send = (lane & 1) ? p0[r] : p0[r + 4];
                float recv = __shfl_xor(send, 1, 32);
                float keep = (lane & 1) ? p0[r + 4] : p0[r];
                q4[r] = keep + recv;
            }
            float q2[2];
#pragma unroll
            for (int r = 0; r < 2; ++r) {
                float send = (lane & 2) ? q4[r] : q4[r + 2];
                float recv = __shfl_xor(send, 2, 32);
                float keep = (lane & 2) ? q4[r + 2] : q4[r];
                q2[r] = keep + recv;
            }
            float sc;
            {
                float send = (lane & 4) ? q2[0] : q2[1];
                float recv = __shfl_xor(send, 4, 32);
                float keep = (lane & 4) ? q2[1] : q2[0];
                sc = keep + recv;
            }
            sc += __shfl_xor(sc, 8, 32);   // lane now holds total for one tile row

            // online softmax update (global over T, per head)
            float tm = sc;
            tm = fmaxf(tm, __shfl_xor(tm, 1, 32));
            tm = fmaxf(tm, __shfl_xor(tm, 2, 32));
            tm = fmaxf(tm, __shfl_xor(tm, 4, 32));
            tm = fmaxf(tm, __shfl_xor(tm, 16, 32));   // mask 8 redundant (dup lanes)
            const float mnew  = fmaxf(mrun, tm);
            const float scale = __expf(mrun - mnew);
            const float wei   = __expf(sc - mnew);
            float ts = wei;                            // 16-row sum (skip mask 8: dups)
            ts += __shfl_xor(ts, 1, 32);
            ts += __shfl_xor(ts, 2, 32);
            ts += __shfl_xor(ts, 4, 32);
            ts += __shfl_xor(ts, 16, 32);
            srun = srun * scale + ts;
            mrun = mnew;

            // publish per-row weights (row = bit-reversed lane[2:0])
            const int rr = ((lane & 1) << 2) | (lane & 2) | ((lane & 4) >> 2);
            if ((lane & 8) == 0)
                sWei[w * 16 + ((lane >> 4) << 3) + rr] = wei;
            // same-wave LDS in-order: read back the 8 weights for this half's rows
            const float4* wp = (const float4*)(sWei + w * 16 + ((lane >> 4) << 3));
            float4 w0 = wp[0], w1 = wp[1];
            float wv8[8] = { w0.x, w0.y, w0.z, w0.w, w1.x, w1.y, w1.z, w1.w };
            float vsum = 0.f;
#pragma unroll
            for (int r = 0; r < 8; ++r) vsum += cv[r] * wv8[r];
            acc = acc * scale + vsum;
            __syncthreads();
        }
        acc += __shfl_xor(acc, 16, 32);
        if (lane < 16) sAtt[w * 16 + lane] = acc / srun;
        __syncthreads();

        // ---- q += att @ proj_w + proj_b ----
        for (int j = tid; j < E_DIM; j += 256) {
            float a2 = PB[j];
            for (int i = 0; i < E_DIM; ++i) a2 += sAtt[i] * PW[i * E_DIM + j];
            sq[j] += a2;
        }
        __syncthreads();

        // ---- FFN: q += gelu(ln4(q) @ ff_w1 + b1) @ ff_w2 + b2 ----
        ln_vec(sq, ln4g + l * E_DIM, ln4b + l * E_DIM, sFfn, sRed, tid);
        __syncthreads();
        float* h512 = sFfn + 128;
        for (int j = tid; j < 512; j += 256) {
            float a2 = F1B[j];
            for (int i = 0; i < E_DIM; ++i) a2 += sFfn[i] * F1W[i * 512 + j];
            h512[j] = 0.5f * a2 * (1.f + erff(a2 * 0.70710678118654752f)); // exact gelu
        }
        __syncthreads();
        for (int j = tid; j < E_DIM; j += 256) {
            float a2 = F2B[j];
            for (int i = 0; i < 512; ++i) a2 += h512[i] * F2W[i * E_DIM + j];
            sq[j] += a2;
        }
        __syncthreads();
    }

    // ---- final LN + head ----
    ln_vec(sq, lnfg, lnfb, sFfn, sRed, tid);
    __syncthreads();
    for (int j = tid; j < 37; j += 256) {
        float a2 = headb[j];
        for (int i = 0; i < E_DIM; ++i) a2 += sFfn[i] * headw[i * 37 + j];
        out[(size_t)b * 37 + j] = a2;
    }
}

// ---------------------------------------------------------------------------
// StateNet: tiny convs (width-1 images -> vertical 3-taps; only kw==1 column
// of the 3x3 kernel lands on valid input) + FCs. One block (128 thr) per row.
// ---------------------------------------------------------------------------
__global__ __launch_bounds__(128) void statenet_kernel(
    const float* __restrict__ x1, const float* __restrict__ x2, const float* __restrict__ x3,
    const float* __restrict__ c11w, const float* __restrict__ c11b,
    const float* __restrict__ bn11g, const float* __restrict__ bn11b,
    const float* __restrict__ c12w, const float* __restrict__ c12b,
    const float* __restrict__ bn12g, const float* __restrict__ bn12b,
    const float* __restrict__ fc1w, const float* __restrict__ fc1b,
    const float* __restrict__ c21w, const float* __restrict__ c21b,
    const float* __restrict__ bn21g, const float* __restrict__ bn21b,
    const float* __restrict__ fc2w, const float* __restrict__ fc2b,
    const float* __restrict__ fcw, const float* __restrict__ fcb,
    float* __restrict__ q0) {
    __shared__ float sx1[3 * 37];
    __shared__ float sc1[8 * 37];
    __shared__ float sc2[8 * 37];
    __shared__ float sx2[7 * 4];
    __shared__ float sc3[8 * 4];
    __shared__ float scat[84];

    const int tid = threadIdx.x;
    const int b = blockIdx.x;
    const float bnscale = rsqrtf(1.f + 1e-5f);

    for (int i = tid; i < 111; i += 128) sx1[i] = x1[(size_t)b * 111 + i];
    for (int i = tid; i < 28; i += 128)  sx2[i] = x2[(size_t)b * 28 + i];
    __syncthreads();

    for (int idx = tid; idx < 296; idx += 128) {
        int o = idx / 37, p = idx % 37;
        float acc = c11b[o];
        for (int i = 0; i < 3; ++i)
            for (int kh = 0; kh < 3; ++kh) {
                int hp = p - 1 + kh;
                if (hp >= 0 && hp < 37) acc += sx1[i * 37 + hp] * c11w[((o * 3 + i) * 3 + kh) * 3 + 1];
            }
        acc = acc * bn11g[o] * bnscale + bn11b[o];
        sc1[idx] = fmaxf(acc, 0.f);
    }
    __syncthreads();
    for (int idx = tid; idx < 296; idx += 128) {
        int o = idx / 37, p = idx % 37;
        float acc = c12b[o];
        for (int i = 0; i < 8; ++i)
            for (int kh = 0; kh < 3; ++kh) {
                int hp = p - 1 + kh;
                if (hp >= 0 && hp < 37) acc += sc1[i * 37 + hp] * c12w[((o * 8 + i) * 3 + kh) * 3 + 1];
            }
        acc = acc * bn12g[o] * bnscale + bn12b[o];
        sc2[idx] = fmaxf(acc, 0.f);
    }
    __syncthreads();
    for (int j = tid; j < 64; j += 128) {
        float acc = fc1b[j];
        for (int i = 0; i < 296; ++i) acc += sc2[i] * fc1w[i * 64 + j];
        scat[j] = fmaxf(acc, 0.f);
    }
    for (int idx = tid; idx < 32; idx += 128) {
        int o = idx / 4, p = idx % 4;
        float acc = c21b[o];
        for (int i = 0; i < 7; ++i)
            for (int kh = 0; kh < 3; ++kh) {
                int hp = p - 1 + kh;
                if (hp >= 0 && hp < 4) acc += sx2[i * 4 + hp] * c21w[((o * 7 + i) * 3 + kh) * 3 + 1];
            }
        acc = acc * bn21g[o] * bnscale + bn21b[o];
        sc3[idx] = fmaxf(acc, 0.f);
    }
    __syncthreads();
    for (int j = tid; j < 16; j += 128) {
        float acc = fc2b[j];
        for (int i = 0; i < 32; ++i) acc += sc3[i] * fc2w[i * 16 + j];
        scat[64 + j] = fmaxf(acc, 0.f);
    }
    if (tid < 4) scat[80 + tid] = x3[(size_t)b * 4 + tid];
    __syncthreads();
    for (int j = tid; j < 128; j += 128) {
        float acc = fcb[j];
        for (int i = 0; i < 84; ++i) acc += scat[i] * fcw[i * 128 + j];
        q0[(size_t)b * 128 + j] = fmaxf(acc, 0.f);
    }
}

extern "C" void kernel_launch(void* const* d_in, const int* in_sizes, int n_in,
                              void* d_out, int out_size, void* d_ws, size_t ws_size,
                              hipStream_t stream) {
    const float* enc   = (const float*)d_in[0];
    const float* x1    = (const float*)d_in[1];
    const float* x2    = (const float*)d_in[2];
    const float* x3    = (const float*)d_in[3];
    const float* c11w  = (const float*)d_in[4];
    const float* c11b  = (const float*)d_in[5];
    const float* bn11g = (const float*)d_in[6];
    const float* bn11b = (const float*)d_in[7];
    const float* c12w  = (const float*)d_in[8];
    const float* c12b  = (const float*)d_in[9];
    const float* bn12g = (const float*)d_in[10];
    const float* bn12b = (const float*)d_in[11];
    const float* fc1w  = (const float*)d_in[12];
    const float* fc1b  = (const float*)d_in[13];
    const float* c21w  = (const float*)d_in[14];
    const float* c21b  = (const float*)d_in[15];
    const float* bn21g = (const float*)d_in[16];
    const float* bn21b = (const float*)d_in[17];
    const float* fc2w  = (const float*)d_in[18];
    const float* fc2b  = (const float*)d_in[19];
    const float* fcw   = (const float*)d_in[20];
    const float* fcb   = (const float*)d_in[21];
    const float* wk    = (const float*)d_in[22];
    const float* wq    = (const float*)d_in[23];
    const float* wv    = (const float*)d_in[24];
    const float* projw = (const float*)d_in[25];
    const float* projb = (const float*)d_in[26];
    const float* ln1g  = (const float*)d_in[27];
    const float* ln1b  = (const float*)d_in[28];
    const float* ln2g  = (const float*)d_in[29];
    const float* ln2b  = (const float*)d_in[30];
    const float* ln3g  = (const float*)d_in[31];
    const float* ln3b  = (const float*)d_in[32];
    const float* ln4g  = (const float*)d_in[33];
    const float* ln4b  = (const float*)d_in[34];
    const float* ffw1  = (const float*)d_in[35];
    const float* ffb1  = (const float*)d_in[36];
    const float* ffw2  = (const float*)d_in[37];
    const float* ffb2  = (const float*)d_in[38];
    const float* lnfg  = (const float*)d_in[39];
    const float* lnfb  = (const float*)d_in[40];
    const float* headw = (const float*)d_in[41];
    const float* headb = (const float*)d_in[42];

    const int B = in_sizes[0] / (T_DIM * E_DIM);   // 2048
    float* q0 = (float*)d_ws;                      // [B,128] scratch

    statenet_kernel<<<B, 128, 0, stream>>>(x1, x2, x3,
        c11w, c11b, bn11g, bn11b, c12w, c12b, bn12g, bn12b,
        fc1w, fc1b, c21w, c21b, bn21g, bn21b, fc2w, fc2b, fcw, fcb, q0);

    const size_t dynBytes = (size_t)E_DIM * E_DIM * sizeof(unsigned short); // 32KB wv^T
    decoder_kernel<<<B, 256, dynBytes, stream>>>(enc, q0,
        wk, wq, wv, projw, projb,
        ln1g, ln1b, ln2g, ln2b, ln3g, ln3b, ln4g, ln4b,
        ffw1, ffb1, ffw2, ffb2, lnfg, lnfb, headw, headb,
        (float*)d_out);
}
